// TransformerRoPE_82686710383441
// MI455X (gfx1250) — compile-verified, matrix operand order checked by reading, not run on hardware
//
#include <hip/hip_runtime.h>
#include <hip/hip_bf16.h>

// ---------------------------------------------------------------------------
// Transformer block (pre-LN RoPE MHA + SwiGLU FFN) for gfx1250 (MI455X).
// All GEMMs use v_wmma_f32_16x16x32_f16 (f16 inputs, f32 accumulation).
// B=4, L=1024, C=1024, NH=16, HD=64.
// ---------------------------------------------------------------------------

typedef __attribute__((ext_vector_type(16))) _Float16 v16h;
typedef __attribute__((ext_vector_type(8)))  _Float16 v8h;
typedef __attribute__((ext_vector_type(8)))  float    v8f;

#define BB 4
#define LL 1024
#define CC 1024
#define NHH 16
#define HDD 64

// ---------------- WMMA helpers -------------------------------------------

__device__ __forceinline__ v8f wmma16(v16h a, v16h b, v8f c) {
  // 8 args: (neg_a, A, neg_b, B, c_mod, C, reuse_a, reuse_b)
  return __builtin_amdgcn_wmma_f32_16x16x32_f16(
      false, a, false, b, (short)0, c, false, false);
}

// A-fragment (16x32, f16) of a row-major [rows, ld] matrix at (row0, k0).
// Per ISA 7.12.2: lanes 0-15 hold M=lane, K = k0+[0..7] and k0+[16..23];
// lanes 16-31 hold M=lane-16, K = k0+[8..15] and k0+[24..31].
// Both chunks are 16B contiguous -> two global_load_b128 per lane.
// B-fragment (32x16) of W stored [N,K] row-major uses the SAME loader with
// row0 = n0 (lane holds column n = lane&15, K chunks contiguous in W's row).
__device__ __forceinline__ v16h load_frag(const _Float16* __restrict__ p,
                                          int ld, int row0, int k0) {
  const int lane = threadIdx.x & 31;
  const int m = lane & 15;
  const int g = lane >> 4;
  const _Float16* r = p + (size_t)(row0 + m) * (size_t)ld + k0 + g * 8;
  v8h lo = *(const v8h*)(r);
  v8h hi = *(const v8h*)(r + 16);
  v16h out;
#pragma unroll
  for (int i = 0; i < 8; ++i) { out[i] = lo[i]; out[i + 8] = hi[i]; }
  return out;
}

// ---------------- elementwise helpers ------------------------------------

__device__ __forceinline__ float redmax16(float v) {
#pragma unroll
  for (int m = 1; m < 16; m <<= 1) v = fmaxf(v, __shfl_xor(v, m, 32));
  return v;
}
__device__ __forceinline__ float redsum16(float v) {
#pragma unroll
  for (int m = 1; m < 16; m <<= 1) v += __shfl_xor(v, m, 32);
  return v;
}

// ---------------- fp32 -> f16 weight conversion --------------------------

__global__ void __launch_bounds__(256) cvt_f32_f16(const float* __restrict__ in,
                                                   _Float16* __restrict__ out,
                                                   int n) {
  int i = blockIdx.x * 256 + threadIdx.x;
  if (i < n) out[i] = (_Float16)in[i];
}

// ---------------- LayerNorm (row of 1024) -> f16 --------------------------

__global__ void __launch_bounds__(256) ln_kernel(const float* __restrict__ x,
                                                 const float* __restrict__ w,
                                                 const float* __restrict__ b,
                                                 _Float16* __restrict__ out) {
  __shared__ float s_sum[8];
  __shared__ float s_sq[8];
  const int row = blockIdx.x;
  const int tid = threadIdx.x;
  const int wave = tid >> 5, lane = tid & 31;
  const float* xr = x + (size_t)row * CC;

  float vals[4];
  float lsum = 0.f, lsq = 0.f;
#pragma unroll
  for (int i = 0; i < 4; ++i) {
    float v = xr[tid + i * 256];
    vals[i] = v;
    lsum += v;
    lsq += v * v;
  }
#pragma unroll
  for (int m = 1; m < 32; m <<= 1) {
    lsum += __shfl_xor(lsum, m, 32);
    lsq  += __shfl_xor(lsq,  m, 32);
  }
  if (lane == 0) { s_sum[wave] = lsum; s_sq[wave] = lsq; }
  __syncthreads();
  float tot = 0.f, totq = 0.f;
#pragma unroll
  for (int i = 0; i < 8; ++i) { tot += s_sum[i]; totq += s_sq[i]; }
  const float mean = tot * (1.0f / CC);
  const float var  = totq * (1.0f / CC) - mean * mean;
  const float rstd = rsqrtf(var + 1e-5f);
#pragma unroll
  for (int i = 0; i < 4; ++i) {
    int c = tid + i * 256;
    out[(size_t)row * CC + c] =
        (_Float16)((vals[i] - mean) * rstd * w[c] + b[c]);
  }
}

// ---------------- Generic GEMM:  out[M,N] = A[M,K] @ W[N,K]^T ------------
// 8 waves, wave tile 64x32, macro tile 128x128. grid = (N/128, M/128).

template <bool F16OUT>
__global__ void __launch_bounds__(256)
gemm_kernel(const _Float16* __restrict__ A, int lda,
            const _Float16* __restrict__ W, int ldw,
            const float* __restrict__ bias,
            const float* __restrict__ addend,
            void* __restrict__ out, int ldo, int K) {
  const int wave = threadIdx.x >> 5;
  const int lane = threadIdx.x & 31;
  const int m0 = blockIdx.y * 128 + (wave & 1) * 64;
  const int n0 = blockIdx.x * 128 + (wave >> 1) * 32;

  v8f acc[4][2];
#pragma unroll
  for (int i = 0; i < 4; ++i)
#pragma unroll
    for (int j = 0; j < 2; ++j)
#pragma unroll
      for (int r = 0; r < 8; ++r) acc[i][j][r] = 0.f;

  for (int k = 0; k < K; k += 32) {
    // L2-resident operands; hint the next K-chunk.
    __builtin_prefetch(A + (size_t)m0 * lda + k + 32, 0, 1);
    __builtin_prefetch(W + (size_t)n0 * ldw + k + 32, 0, 1);
    v16h a[4], b[2];
#pragma unroll
    for (int i = 0; i < 4; ++i) a[i] = load_frag(A, lda, m0 + i * 16, k);
#pragma unroll
    for (int j = 0; j < 2; ++j) b[j] = load_frag(W, ldw, n0 + j * 16, k);
#pragma unroll
    for (int i = 0; i < 4; ++i)
#pragma unroll
      for (int j = 0; j < 2; ++j) acc[i][j] = wmma16(a[i], b[j], acc[i][j]);
  }

  const int col = lane & 15, g = lane >> 4;
#pragma unroll
  for (int i = 0; i < 4; ++i)
#pragma unroll
    for (int j = 0; j < 2; ++j) {
      const int mbase = m0 + i * 16 + 8 * g;
      const int nn = n0 + j * 16 + col;
      const float bs = bias ? bias[nn] : 0.f;
#pragma unroll
      for (int r = 0; r < 8; ++r) {
        const size_t idx = (size_t)(mbase + r) * (size_t)ldo + nn;
        float v = acc[i][j][r] + bs;
        if (addend) v += addend[idx];
        if (F16OUT) ((_Float16*)out)[idx] = (_Float16)v;
        else        ((float*)out)[idx] = v;
      }
    }
}

// ---------------- SwiGLU GEMM ---------------------------------------------
// gates = h2 @ sw_w^T + sw_b ; act = g0 * sigmoid(beta*g1) * g1
// Each wave computes the (m,n) tile of g0 AND the (m,n+4C) tile of g1 so the
// activation fuses in-register. Wave tile 32x32 (dual), 8 waves as 4(M)x2(N)
// -> macro 128x64. grid = (4096/64, 4096/128).

__global__ void __launch_bounds__(256)
swiglu_gemm(const _Float16* __restrict__ A,       // [4096,1024]
            const _Float16* __restrict__ W,       // [8192,1024]
            const float* __restrict__ swb,        // [8192]
            const float* __restrict__ beta,       // [4096]
            _Float16* __restrict__ act) {         // [4096,4096]
  const int wave = threadIdx.x >> 5;
  const int lane = threadIdx.x & 31;
  const int m0 = blockIdx.y * 128 + (wave >> 1) * 32;
  const int n0 = blockIdx.x * 64 + (wave & 1) * 32;

  v8f acc0[2][2], acc1[2][2];
#pragma unroll
  for (int i = 0; i < 2; ++i)
#pragma unroll
    for (int j = 0; j < 2; ++j)
#pragma unroll
      for (int r = 0; r < 8; ++r) { acc0[i][j][r] = 0.f; acc1[i][j][r] = 0.f; }

  for (int k = 0; k < CC; k += 32) {
    v16h a[2], b0[2], b1[2];
#pragma unroll
    for (int i = 0; i < 2; ++i) a[i] = load_frag(A, CC, m0 + i * 16, k);
#pragma unroll
    for (int j = 0; j < 2; ++j) {
      b0[j] = load_frag(W, CC, n0 + j * 16, k);
      b1[j] = load_frag(W, CC, 4096 + n0 + j * 16, k);
    }
#pragma unroll
    for (int i = 0; i < 2; ++i)
#pragma unroll
      for (int j = 0; j < 2; ++j) {
        acc0[i][j] = wmma16(a[i], b0[j], acc0[i][j]);
        acc1[i][j] = wmma16(a[i], b1[j], acc1[i][j]);
      }
  }

  const int col = lane & 15, g = lane >> 4;
#pragma unroll
  for (int i = 0; i < 2; ++i)
#pragma unroll
    for (int j = 0; j < 2; ++j) {
      const int mbase = m0 + i * 16 + 8 * g;
      const int nn = n0 + j * 16 + col;
      const float b0s = swb[nn];
      const float b1s = swb[4096 + nn];
      const float bet = beta[nn];
#pragma unroll
      for (int r = 0; r < 8; ++r) {
        float g0v = acc0[i][j][r] + b0s;
        float g1v = acc1[i][j][r] + b1s;
        float sig = 1.f / (1.f + __expf(-bet * g1v));
        act[(size_t)(mbase + r) * 4096 + nn] = (_Float16)(g0v * sig * g1v);
      }
    }
}

// ---------------- RoPE + head split + V transpose -------------------------
// kqv16 chunk order along 3C is k, q, v (matches jnp.split in the reference).

__global__ void __launch_bounds__(256)
rope_kernel(const _Float16* __restrict__ kqv,   // [B*L, 3C] f16
            _Float16* __restrict__ q16,         // [B,NH,L,HD]
            _Float16* __restrict__ k16,         // [B,NH,L,HD]
            _Float16* __restrict__ vt16) {      // [B,NH,HD,L]
  const int tid = blockIdx.x * 256 + threadIdx.x;   // B*NH*L*32 threads
  const int j = tid & 31;                // pair index (d = 2j, 2j+1)
  const int l = (tid >> 5) & 1023;
  const int h = (tid >> 15) & 15;
  const int b = tid >> 19;

  const _Float16* row = kqv + (size_t)(b * LL + l) * (3 * CC);
  const int ch = h * HDD + 2 * j;
  const float k0v = (float)row[ch],            k1v = (float)row[ch + 1];
  const float q0v = (float)row[CC + ch],       q1v = (float)row[CC + ch + 1];
  const float v0v = (float)row[2 * CC + ch],   v1v = (float)row[2 * CC + ch + 1];

  const float inv = powf(10000.f, -(float)(2 * j) / (float)HDD);
  const float fr = (float)l * inv;
  const float c = cosf(fr), s = sinf(fr);

  const size_t qb = ((size_t)((b * NHH + h) * LL + l)) * HDD + 2 * j;
  q16[qb]     = (_Float16)(q0v * c - q1v * s);
  q16[qb + 1] = (_Float16)(q1v * c + q0v * s);
  k16[qb]     = (_Float16)(k0v * c - k1v * s);
  k16[qb + 1] = (_Float16)(k1v * c + k0v * s);

  const size_t vb = ((size_t)((b * NHH + h) * HDD + 2 * j)) * LL + l;
  vt16[vb]      = (_Float16)v0v;
  vt16[vb + LL] = (_Float16)v1v;
}

// ---------------- Flash attention ----------------------------------------
// One wave handles 16 query rows of one (b,h). scores = q k^T * sqrt(HD),
// streaming softmax, o = P v via LDS A-layout transpose of P.
// grid.x = B*NH*(L/16)/8, block = 256 (8 waves), wave-private LDS tiles.

__global__ void __launch_bounds__(256)
attn_kernel(const _Float16* __restrict__ q,    // [B,NH,L,HD]
            const _Float16* __restrict__ k,    // [B,NH,L,HD]
            const _Float16* __restrict__ vt,   // [B,NH,HD,L]
            _Float16* __restrict__ o) {        // [B,L,C]
  __shared__ __align__(16) _Float16 plds[8][16 * 32];

  const int wave = threadIdx.x >> 5;
  const int lane = threadIdx.x & 31;
  const int gw = blockIdx.x * 8 + wave;   // 0 .. B*NH*64-1
  const int mt = gw & 63;
  const int bh = gw >> 6;
  const int b = bh >> 4, h = bh & 15;
  const int m0 = mt * 16;

  const _Float16* qb = q + (size_t)bh * LL * HDD;
  const _Float16* kb = k + (size_t)bh * LL * HDD;
  const _Float16* vb = vt + (size_t)bh * HDD * LL;

  const v16h a0 = load_frag(qb, HDD, m0, 0);
  const v16h a1 = load_frag(qb, HDD, m0, 32);

  v8f oacc[4];
#pragma unroll
  for (int dt = 0; dt < 4; ++dt)
#pragma unroll
    for (int r = 0; r < 8; ++r) oacc[dt][r] = 0.f;

  float mrow[8], lrow[8];
#pragma unroll
  for (int r = 0; r < 8; ++r) { mrow[r] = -1e30f; lrow[r] = 0.f; }

  const int col = lane & 15, g = lane >> 4;
  _Float16* pl = plds[wave];

  for (int nt = 0; nt < 64; nt += 2) {     // 32 keys per chunk
    v8f s[2];
#pragma unroll
    for (int t = 0; t < 2; ++t) {
      v16h b0 = load_frag(kb, HDD, (nt + t) * 16, 0);
      v16h b1 = load_frag(kb, HDD, (nt + t) * 16, 32);
      v8f z;
#pragma unroll
      for (int r = 0; r < 8; ++r) z[r] = 0.f;
      z = wmma16(a0, b0, z);
      z = wmma16(a1, b1, z);
#pragma unroll
      for (int r = 0; r < 8; ++r) z[r] *= 8.0f;   // * sqrt(HD)
      s[t] = z;
    }

    // streaming softmax update (rows r+8*g live across 16-lane halves)
    float scale[8];
#pragma unroll
    for (int r = 0; r < 8; ++r) {
      float tm = redmax16(fmaxf(s[0][r], s[1][r]));
      float nm = fmaxf(mrow[r], tm);
      scale[r] = __expf(mrow[r] - nm);
      mrow[r] = nm;
      s[0][r] = __expf(s[0][r] - nm);
      s[1][r] = __expf(s[1][r] - nm);
      float ts = redsum16(s[0][r] + s[1][r]);
      lrow[r] = lrow[r] * scale[r] + ts;
    }
#pragma unroll
    for (int dt = 0; dt < 4; ++dt)
#pragma unroll
      for (int r = 0; r < 8; ++r) oacc[dt][r] *= scale[r];

    // write P (C-layout) to wave-private LDS, reload as A-fragment
#pragma unroll
    for (int r = 0; r < 8; ++r) {
      pl[(8 * g + r) * 32 + col]      = (_Float16)s[0][r];
      pl[(8 * g + r) * 32 + 16 + col] = (_Float16)s[1][r];
    }
    __syncthreads();   // uniform across the block; orders LDS store->load
    const v16h ap = load_frag(pl, 32, 0, 0);

#pragma unroll
    for (int dt = 0; dt < 4; ++dt) {
      v16h bv = load_frag(vb, LL, dt * 16, nt * 16);
      oacc[dt] = wmma16(ap, bv, oacc[dt]);
    }
    __syncthreads();
  }

  // normalize + store o[b, l, h*64 + d]
#pragma unroll
  for (int dt = 0; dt < 4; ++dt)
#pragma unroll
    for (int r = 0; r < 8; ++r) {
      const int l = m0 + 8 * g + r;
      const float val = oacc[dt][r] / lrow[r];
      o[((size_t)(b * LL + l)) * CC + h * HDD + dt * 16 + col] = (_Float16)val;
    }
}

// ---------------- Launcher -------------------------------------------------

extern "C" void kernel_launch(void* const* d_in, const int* in_sizes, int n_in,
                              void* d_out, int out_size, void* d_ws, size_t ws_size,
                              hipStream_t stream) {
  (void)in_sizes; (void)n_in; (void)out_size; (void)ws_size;

  const float* x     = (const float*)d_in[0];
  const float* ln1_w = (const float*)d_in[1];
  const float* ln1_b = (const float*)d_in[2];
  const float* ln2_w = (const float*)d_in[3];
  const float* ln2_b = (const float*)d_in[4];
  const float* kqv_w = (const float*)d_in[5];
  const float* out_w = (const float*)d_in[6];
  const float* out_b = (const float*)d_in[7];
  const float* sw_w  = (const float*)d_in[8];
  const float* sw_b  = (const float*)d_in[9];
  const float* beta  = (const float*)d_in[10];
  const float* ffn_w = (const float*)d_in[11];
  const float* ffn_b = (const float*)d_in[12];
  float* outp = (float*)d_out;

  char* ws = (char*)d_ws;
  // workspace layout (bytes), lifetimes overlaid:
  _Float16* wkqv = (_Float16*)(ws + 0);                    // 6 MiB  [3072,1024]
  _Float16* wout = (_Float16*)(ws + 6291456);              // 2 MiB  [1024,1024]
  _Float16* wsw  = (_Float16*)(ws + 8388608);              // 16 MiB [8192,1024]
  _Float16* wffn = (_Float16*)(ws + 25165824);             // 8 MiB  [1024,4096]
  _Float16* h16  = (_Float16*)(ws + 33554432);             // 8 MiB  (h then h2)
  _Float16* q16  = (_Float16*)(ws + 41943040);             // 8 MiB
  _Float16* k16  = (_Float16*)(ws + 50331648);             // 8 MiB
  _Float16* vt16 = (_Float16*)(ws + 58720256);             // 8 MiB
  _Float16* kqv16 = (_Float16*)(ws + 67108864);            // 24 MiB [4096,3072]
  _Float16* o16   = (_Float16*)(ws + 67108864);            // 8 MiB (after rope)
  float*    resid = (float*)  (ws + 75497472);             // 16 MiB [4096,1024]
  _Float16* act16 = (_Float16*)(ws + 41943040);            // 32 MiB (after attn)
  // total required: 92274688 bytes (88 MiB)

  const int M = BB * LL;  // 4096

  // 1. weights fp32 -> f16
  cvt_f32_f16<<<(3072 * 1024) / 256, 256, 0, stream>>>(kqv_w, wkqv, 3072 * 1024);
  cvt_f32_f16<<<(1024 * 1024) / 256, 256, 0, stream>>>(out_w, wout, 1024 * 1024);
  cvt_f32_f16<<<(8192 * 1024) / 256, 256, 0, stream>>>(sw_w,  wsw,  8192 * 1024);
  cvt_f32_f16<<<(1024 * 4096) / 256, 256, 0, stream>>>(ffn_w, wffn, 1024 * 4096);

  // 2. h = LN1(x) -> f16
  ln_kernel<<<M, 256, 0, stream>>>(x, ln1_w, ln1_b, h16);

  // 3. kqv = h @ kqv_w^T  -> f16 [4096,3072]
  gemm_kernel<true><<<dim3(3072 / 128, M / 128), 256, 0, stream>>>(
      h16, CC, wkqv, CC, nullptr, nullptr, (void*)kqv16, 3072, CC);

  // 4. RoPE + head split + V transpose
  rope_kernel<<<(BB * NHH * LL * 32) / 256, 256, 0, stream>>>(kqv16, q16, k16, vt16);

  // 5. attention -> o16 [4096,1024] (overlays kqv16, now dead)
  attn_kernel<<<(BB * NHH * (LL / 16)) / 8, 256, 0, stream>>>(q16, k16, vt16, o16);

  // 6. resid = o @ out_w^T + out_b + x  -> f32
  gemm_kernel<false><<<dim3(CC / 128, M / 128), 256, 0, stream>>>(
      o16, CC, wout, CC, out_b, x, (void*)resid, CC, CC);

  // 7. h2 = LN2(resid) -> f16 (reuses h16)
  ln_kernel<<<M, 256, 0, stream>>>(resid, ln2_w, ln2_b, h16);

  // 8. fused SwiGLU GEMM -> act16 [4096,4096] (overlays rope bufs + o16)
  swiglu_gemm<<<dim3(4096 / 64, M / 128), 256, 0, stream>>>(
      h16, wsw, sw_b, beta, act16);

  // 9. out = act @ ffn_w^T + ffn_b + resid -> f32 d_out
  gemm_kernel<false><<<dim3(CC / 128, M / 128), 256, 0, stream>>>(
      act16, 4096, wffn, 4096, ffn_b, resid, (void*)outp, CC, 4096);
}
